// AngularAwareTemporalAttention_73607149519417
// MI455X (gfx1250) — matrix operand order repaired and checked by reading, not compiled, not run on hardware
//
#include <hip/hip_runtime.h>
#include <math.h>

typedef __bf16 bf16;
typedef __attribute__((ext_vector_type(16))) __bf16 v16bf;
typedef __attribute__((ext_vector_type(4)))  __bf16 v4bf;
typedef __attribute__((ext_vector_type(8)))  float  v8f;
typedef __attribute__((ext_vector_type(4)))  int    v4i;

#define D_HID   1024
#define N_HEADS 16
#define D_HEAD  64
#define SEQ     128
#define N_BN    256              /* B*N = 4*64 */
#define N_BATCH 4
#define M_TOT   (N_BN * SEQ)     /* 32768 */
#define N_QKV   (3 * D_HID)      /* 3072 */
#define ATTN_SCALE 0.125f        /* 64^-0.5 */
#define KTILE   64
#define NKT     (D_HID / KTILE)  /* 16 */

// ---------------------------------------------------------------------------
// CDNA5 async global->LDS copy (ASYNCcnt path) with synchronous fallback
// Builtin signature (from hipcc diagnostic): param0 = v4i in global AS(1),
// param1 = v4i in LDS AS(3), then imm offset, imm cpol.
// ---------------------------------------------------------------------------
#if __has_builtin(__builtin_amdgcn_global_load_async_to_lds_b128)
#define HAVE_ASYNC_COPY 1
typedef __attribute__((address_space(1))) v4i gas_v4i;
typedef __attribute__((address_space(3))) v4i las_v4i;
#else
#define HAVE_ASYNC_COPY 0
#endif

__device__ __forceinline__ void lds_copy16(void* lds_dst, const void* gsrc) {
#if HAVE_ASYNC_COPY
  __builtin_amdgcn_global_load_async_to_lds_b128(
      (gas_v4i*)gsrc, (las_v4i*)lds_dst, 0, 0);
#else
  *reinterpret_cast<int4*>(lds_dst) = *reinterpret_cast<const int4*>(gsrc);
#endif
}

__device__ __forceinline__ void wait_async() {
#if HAVE_ASYNC_COPY
#if __has_builtin(__builtin_amdgcn_s_wait_asynccnt)
  __builtin_amdgcn_s_wait_asynccnt(0);
#else
  asm volatile("s_wait_asynccnt 0" ::: "memory");
#endif
#endif
}

// ---------------------------------------------------------------------------
// WMMA helpers (CDNA5 wave32, 16x16x32 bf16 -> f32)
// ---------------------------------------------------------------------------
__device__ __forceinline__ v8f wmma_bf16(const v16bf a, const v16bf b, v8f c) {
  // 8 args: (neg_a, A, neg_b, B, c_mod, C, reuse_a, reuse_b)
  return __builtin_amdgcn_wmma_f32_16x16x32_bf16(false, a, false, b, (short)0, c,
                                                 false, false);
}

// A fragment: 16x32 bf16 tile, row-major in LDS with leading dim `ld`.
// Lane L(0..15)=row M=L; lane-half kh in {0,8}; two contiguous 8-element runs.
__device__ __forceinline__ v16bf load_a_frag(const bf16* base, int ld, int mbase,
                                             int kbase, int lane) {
  const bf16* p = base + (size_t)(mbase + (lane & 15)) * ld + kbase;
  const int kh = (lane >> 4) * 8;
  v16bf a;
#pragma unroll
  for (int j = 0; j < 8; ++j) a[j] = p[kh + j];
#pragma unroll
  for (int j = 0; j < 8; ++j) a[j + 8] = p[16 + kh + j];
  return a;
}

// B fragment: 32x16 tile stored column-major in LDS ([n][k], leading dim `ld`).
// Lane L(0..15)=col N=L; lane-half kh in {0,8}; two contiguous 8-element runs.
__device__ __forceinline__ v16bf load_b_frag(const bf16* base, int ld, int nbase,
                                             int kbase, int lane) {
  const bf16* p = base + (size_t)(nbase + (lane & 15)) * ld + kbase;
  const int kh = (lane >> 4) * 8;
  v16bf b;
#pragma unroll
  for (int j = 0; j < 8; ++j) b[j] = p[kh + j];
#pragma unroll
  for (int j = 0; j < 8; ++j) b[j + 8] = p[16 + kh + j];
  return b;
}

// ---------------------------------------------------------------------------
// weight prep: fp32 (K x N) row-major -> bf16 (N x K) row-major (transposed)
// so GEMM B tiles become row-contiguous 16-byte async copies.
// ---------------------------------------------------------------------------
__global__ void cvt_transpose_kernel(const float* __restrict__ src,
                                     bf16* __restrict__ dst, int ncols) {
  const int total = D_HID * ncols;
  for (int i = blockIdx.x * blockDim.x + threadIdx.x; i < total;
       i += gridDim.x * blockDim.x) {
    const int n = i >> 10;              // row of dst (N dim)
    const int k = i & (D_HID - 1);      // col of dst (K dim), K = 1024
    dst[i] = (bf16)src[(size_t)k * ncols + n];
  }
}

// ---------------------------------------------------------------------------
// angular bias: normalize bvecs, cos-sim, clip, pre-multiply by scalar scale
// ---------------------------------------------------------------------------
__global__ __launch_bounds__(SEQ) void bias_kernel(const float* __restrict__ bvecs,
                                                   const float* __restrict__ scale_p,
                                                   float* __restrict__ bias_out) {
  __shared__ float nb[SEQ][3];
  const int b = blockIdx.x, t = threadIdx.x;
  const float vx = bvecs[(b * SEQ + t) * 3 + 0];
  const float vy = bvecs[(b * SEQ + t) * 3 + 1];
  const float vz = bvecs[(b * SEQ + t) * 3 + 2];
  const float inv = 1.f / (sqrtf(vx * vx + vy * vy + vz * vz) + 1e-6f);
  nb[t][0] = vx * inv; nb[t][1] = vy * inv; nb[t][2] = vz * inv;
  __syncthreads();
  const float s = scale_p[0];
  const float xi = nb[t][0], yi = nb[t][1], zi = nb[t][2];
  for (int j = 0; j < SEQ; ++j) {
    float d = xi * nb[j][0] + yi * nb[j][1] + zi * nb[j][2];
    d = fminf(1.f, fmaxf(-1.f, d));
    bias_out[((size_t)b * SEQ + t) * SEQ + j] = d * s;
  }
}

// ---------------------------------------------------------------------------
// QKV GEMM: (32768 x 1024 f32) x W^T(3072 x 1024 bf16) + qkv_b
// -> scatter bf16 into 3 x (BN, H, SEQ, D_HEAD)
// 128x128 tile, double-buffered KTILE=64, 8 waves, wave = 2x4 WMMA tiles.
// ---------------------------------------------------------------------------
__global__ __launch_bounds__(256) void qkv_gemm_kernel(
    const float* __restrict__ x, const bf16* __restrict__ wt,
    const float* __restrict__ qkv_b, bf16* __restrict__ qkv_out) {
  __shared__ bf16 As[2][128][KTILE];   // [m][k]
  __shared__ bf16 Bs[2][128][KTILE];   // [n][k]
  const int tid = threadIdx.x;
  const int lane = tid & 31, wid = tid >> 5;
  const int wrow = wid & 3, wcol = wid >> 2;
  const int mbase = blockIdx.y * 128;
  const int nbase = blockIdx.x * 128;

  auto stage = [&](int buf, int k0) {
    // A: fp32 -> bf16 through VGPRs (float4 load -> packed bf16x4 LDS store)
    for (int i = tid * 4; i < 128 * KTILE; i += 256 * 4) {
      const int m = i >> 6, kk = i & (KTILE - 1);
      const float4 f = *(const float4*)&x[(size_t)(mbase + m) * D_HID + k0 + kk];
      v4bf c;
      c[0] = (bf16)f.x; c[1] = (bf16)f.y; c[2] = (bf16)f.z; c[3] = (bf16)f.w;
      *(v4bf*)&As[buf][m][kk] = c;
    }
    // B: row-contiguous 16B chunks, async global->LDS
    for (int c = tid; c < 128 * KTILE / 8; c += 256) {
      const int n = c >> 3, koff = (c & 7) * 8;
      lds_copy16(&Bs[buf][n][koff], &wt[(size_t)(nbase + n) * D_HID + k0 + koff]);
    }
  };

  v8f acc[2][4] = {};
  stage(0, 0);
  for (int kt = 0; kt < NKT; ++kt) {
    const int cur = kt & 1;
    wait_async();
    __syncthreads();
    if (kt + 1 < NKT) stage(1 - cur, (kt + 1) * KTILE);
#pragma unroll
    for (int kc = 0; kc < KTILE; kc += 32) {
      const v16bf a0 = load_a_frag(&As[cur][0][0], KTILE, wrow * 32, kc, lane);
      const v16bf a1 = load_a_frag(&As[cur][0][0], KTILE, wrow * 32 + 16, kc, lane);
#pragma unroll
      for (int j = 0; j < 4; ++j) {
        const v16bf b = load_b_frag(&Bs[cur][0][0], KTILE, wcol * 64 + j * 16, kc, lane);
        acc[0][j] = wmma_bf16(a0, b, acc[0][j]);
        acc[1][j] = wmma_bf16(a1, b, acc[1][j]);
      }
    }
  }

  const size_t selstride = (size_t)N_BN * N_HEADS * SEQ * D_HEAD;
#pragma unroll
  for (int mi = 0; mi < 2; ++mi) {
#pragma unroll
    for (int j = 0; j < 4; ++j) {
      const int gcol = nbase + wcol * 64 + j * 16 + (lane & 15);
      const int mtop = mbase + wrow * 32 + mi * 16 + (lane >> 4) * 8;
      const int sel = gcol >> 10;
      const int head = (gcol >> 6) & (N_HEADS - 1);
      const int hd = gcol & (D_HEAD - 1);
      const float bia = qkv_b[gcol];
      bf16* dst = qkv_out + (size_t)sel * selstride;
#pragma unroll
      for (int r = 0; r < 8; ++r) {
        const int grow = mtop + r;
        const int bn = grow >> 7, t = grow & (SEQ - 1);
        dst[(((size_t)(bn * N_HEADS + head)) * SEQ + t) * D_HEAD + hd] =
            (bf16)(acc[mi][j][r] + bia);
      }
    }
  }
}

// ---------------------------------------------------------------------------
// Fused attention: one block per (bn, head); S kept entirely in LDS (~116 KB).
// ---------------------------------------------------------------------------
__global__ __launch_bounds__(256) void attn_kernel(
    const bf16* __restrict__ qg, const bf16* __restrict__ kg,
    const bf16* __restrict__ vg, const float* __restrict__ bias,
    bf16* __restrict__ ctx) {
  __shared__ union {
    struct { bf16 q[SEQ][D_HEAD]; bf16 k[SEQ][D_HEAD]; } qk;   // 32 KB
    bf16 p[SEQ][136];                                          // 34 KB (padded)
  } u;
  __shared__ bf16 Vt[D_HEAD][130];   // V transposed [hd][t], padded
  __shared__ float S[SEQ][132];      // scores, padded
  const int h = blockIdx.x, bn = blockIdx.y;
  const int tid = threadIdx.x, lane = tid & 31, wid = tid >> 5;
  const size_t head_off = ((size_t)(bn * N_HEADS + h)) * SEQ * D_HEAD;
  const float* bp = bias + (size_t)(bn >> 6) * SEQ * SEQ;

  // Q/K: async 16B copies; V: transpose through VGPRs
  for (int c = tid; c < SEQ * D_HEAD / 8; c += 256) {
    const int t = c >> 3, koff = (c & 7) * 8;
    lds_copy16(&u.qk.q[t][koff], &qg[head_off + t * D_HEAD + koff]);
    lds_copy16(&u.qk.k[t][koff], &kg[head_off + t * D_HEAD + koff]);
  }
  for (int i = tid; i < SEQ * D_HEAD; i += 256) {
    const int t = i >> 6, d = i & (D_HEAD - 1);
    Vt[d][t] = vg[head_off + i];
  }
  wait_async();
  __syncthreads();

  // ---- S = Q K^T * scale + bias -------------------------------------------
  {
    const int wrow = wid & 3, wcol = wid >> 2;
    v8f acc[2][4] = {};
#pragma unroll
    for (int kc = 0; kc < 2; ++kc) {
      const v16bf a0 = load_a_frag(&u.qk.q[0][0], D_HEAD, wrow * 32, kc * 32, lane);
      const v16bf a1 = load_a_frag(&u.qk.q[0][0], D_HEAD, wrow * 32 + 16, kc * 32, lane);
#pragma unroll
      for (int j = 0; j < 4; ++j) {
        const v16bf b = load_b_frag(&u.qk.k[0][0], D_HEAD, wcol * 64 + j * 16,
                                    kc * 32, lane);
        acc[0][j] = wmma_bf16(a0, b, acc[0][j]);
        acc[1][j] = wmma_bf16(a1, b, acc[1][j]);
      }
    }
#pragma unroll
    for (int mi = 0; mi < 2; ++mi) {
#pragma unroll
      for (int j = 0; j < 4; ++j) {
        const int nn = wcol * 64 + j * 16 + (lane & 15);
        const int mtop = wrow * 32 + mi * 16 + (lane >> 4) * 8;
#pragma unroll
        for (int r = 0; r < 8; ++r) {
          const int m = mtop + r;
          S[m][nn] = acc[mi][j][r] * ATTN_SCALE + bp[m * SEQ + nn];
        }
      }
    }
  }
  __syncthreads();

  // ---- wave-parallel softmax: 16 rows/wave, lane-contiguous, shfl reduce ---
  for (int r = wid * 16; r < wid * 16 + 16; ++r) {
    const float v0 = S[r][lane];
    const float v1 = S[r][lane + 32];
    const float v2 = S[r][lane + 64];
    const float v3 = S[r][lane + 96];
    float mx = fmaxf(fmaxf(v0, v1), fmaxf(v2, v3));
#pragma unroll
    for (int off = 16; off > 0; off >>= 1) mx = fmaxf(mx, __shfl_xor(mx, off, 32));
    const float e0 = __expf(v0 - mx), e1 = __expf(v1 - mx);
    const float e2 = __expf(v2 - mx), e3 = __expf(v3 - mx);
    float sum = (e0 + e1) + (e2 + e3);
#pragma unroll
    for (int off = 16; off > 0; off >>= 1) sum += __shfl_xor(sum, off, 32);
    const float inv = 1.f / sum;
    u.p[r][lane]      = (bf16)(e0 * inv);
    u.p[r][lane + 32] = (bf16)(e1 * inv);
    u.p[r][lane + 64] = (bf16)(e2 * inv);
    u.p[r][lane + 96] = (bf16)(e3 * inv);
  }
  __syncthreads();

  // ---- O = P @ V ----------------------------------------------------------
  {
    v8f acc[4] = {};
#pragma unroll
    for (int kc = 0; kc < 4; ++kc) {
      const v16bf a = load_a_frag(&u.p[0][0], 136, wid * 16, kc * 32, lane);
#pragma unroll
      for (int j = 0; j < 4; ++j) {
        const v16bf b = load_b_frag(&Vt[0][0], 130, j * 16, kc * 32, lane);
        acc[j] = wmma_bf16(a, b, acc[j]);
      }
    }
#pragma unroll
    for (int j = 0; j < 4; ++j) {
      const int nn = j * 16 + (lane & 15);
      const int mtop = wid * 16 + (lane >> 4) * 8;
#pragma unroll
      for (int r = 0; r < 8; ++r) {
        const int m = mtop + r;
        ctx[((size_t)(bn * SEQ + m)) * D_HID + h * D_HEAD + nn] = (bf16)acc[j][r];
      }
    }
  }
}

// ---------------------------------------------------------------------------
// Projection GEMM: (32768 x 1024 bf16) x W^T(1024 x 1024 bf16) + proj_b -> f32
// Both operands staged with async global->LDS copies.
// ---------------------------------------------------------------------------
__global__ __launch_bounds__(256) void proj_gemm_kernel(
    const bf16* __restrict__ a, const bf16* __restrict__ wt,
    const float* __restrict__ pb, float* __restrict__ out) {
  __shared__ bf16 As[2][128][KTILE];
  __shared__ bf16 Bs[2][128][KTILE];
  const int tid = threadIdx.x, lane = tid & 31, wid = tid >> 5;
  const int wrow = wid & 3, wcol = wid >> 2;
  const int mbase = blockIdx.y * 128, nbase = blockIdx.x * 128;

  auto stage = [&](int buf, int k0) {
    for (int c = tid; c < 128 * KTILE / 8; c += 256) {
      const int m = c >> 3, koff = (c & 7) * 8;
      lds_copy16(&As[buf][m][koff], &a[(size_t)(mbase + m) * D_HID + k0 + koff]);
      lds_copy16(&Bs[buf][m][koff], &wt[(size_t)(nbase + m) * D_HID + k0 + koff]);
    }
  };

  v8f acc[2][4] = {};
  stage(0, 0);
  for (int kt = 0; kt < NKT; ++kt) {
    const int cur = kt & 1;
    wait_async();
    __syncthreads();
    if (kt + 1 < NKT) stage(1 - cur, (kt + 1) * KTILE);
#pragma unroll
    for (int kc = 0; kc < KTILE; kc += 32) {
      const v16bf a0 = load_a_frag(&As[cur][0][0], KTILE, wrow * 32, kc, lane);
      const v16bf a1 = load_a_frag(&As[cur][0][0], KTILE, wrow * 32 + 16, kc, lane);
#pragma unroll
      for (int j = 0; j < 4; ++j) {
        const v16bf b = load_b_frag(&Bs[cur][0][0], KTILE, wcol * 64 + j * 16, kc, lane);
        acc[0][j] = wmma_bf16(a0, b, acc[0][j]);
        acc[1][j] = wmma_bf16(a1, b, acc[1][j]);
      }
    }
  }
#pragma unroll
  for (int mi = 0; mi < 2; ++mi) {
#pragma unroll
    for (int j = 0; j < 4; ++j) {
      const int gcol = nbase + wcol * 64 + j * 16 + (lane & 15);
      const int mtop = mbase + wrow * 32 + mi * 16 + (lane >> 4) * 8;
      const float bb = pb[gcol];
#pragma unroll
      for (int r = 0; r < 8; ++r)
        out[(size_t)(mtop + r) * D_HID + gcol] = acc[mi][j][r] + bb;
    }
  }
}

// ---------------------------------------------------------------------------
extern "C" void kernel_launch(void* const* d_in, const int* in_sizes, int n_in,
                              void* d_out, int out_size, void* d_ws, size_t ws_size,
                              hipStream_t stream) {
  (void)in_sizes; (void)n_in; (void)out_size; (void)ws_size;
  const float* x      = (const float*)d_in[0];  // (BN, SEQ, 1024)
  const float* bvecs  = (const float*)d_in[1];  // (B, SEQ, 3)
  const float* qkv_w  = (const float*)d_in[2];  // (1024, 3072)
  const float* qkv_b  = (const float*)d_in[3];  // (3072,)
  const float* proj_w = (const float*)d_in[4];  // (1024, 1024)
  const float* proj_b = (const float*)d_in[5];  // (1024,)
  const float* ascale = (const float*)d_in[6];  // (1,)
  float* out = (float*)d_out;                   // (BN, SEQ, 1024)

  char* ws = (char*)d_ws;
  size_t off = 0;
  auto alloc = [&](size_t bytes) -> char* {
    char* p = ws + off;
    off = (off + bytes + 255) & ~(size_t)255;
    return p;
  };
  bf16*  qkv_wt  = (bf16*)alloc((size_t)D_HID * N_QKV * sizeof(bf16));   // [3072][1024]
  bf16*  proj_wt = (bf16*)alloc((size_t)D_HID * D_HID * sizeof(bf16));   // [1024][1024]
  float* biasb   = (float*)alloc((size_t)N_BATCH * SEQ * SEQ * sizeof(float));
  bf16*  qkvbuf  = (bf16*)alloc((size_t)3 * N_BN * N_HEADS * SEQ * D_HEAD * sizeof(bf16));
  bf16*  ctx     = (bf16*)alloc((size_t)M_TOT * D_HID * sizeof(bf16));

  const size_t head_elems = (size_t)N_BN * N_HEADS * SEQ * D_HEAD;

  cvt_transpose_kernel<<<2048, 256, 0, stream>>>(qkv_w, qkv_wt, N_QKV);
  cvt_transpose_kernel<<<1024, 256, 0, stream>>>(proj_w, proj_wt, D_HID);
  bias_kernel<<<N_BATCH, SEQ, 0, stream>>>(bvecs, ascale, biasb);
  qkv_gemm_kernel<<<dim3(N_QKV / 128, M_TOT / 128), 256, 0, stream>>>(
      x, qkv_wt, qkv_b, qkvbuf);
  attn_kernel<<<dim3(N_HEADS, N_BN), 256, 0, stream>>>(
      qkvbuf, qkvbuf + head_elems, qkvbuf + 2 * head_elems, biasb, ctx);
  proj_gemm_kernel<<<dim3(D_HID / 128, M_TOT / 128), 256, 0, stream>>>(
      ctx, proj_wt, proj_b, out);
}
